// CharacterCountModel_47167330845421
// MI455X (gfx1250) — matrix operand order repaired and checked by reading, not compiled, not run on hardware
//
#include <hip/hip_runtime.h>

// ---------------------------------------------------------------------------
// 2-layer GRU (B=256, T=4096, I=26, H=128) + FC on last step, for gfx1250.
// Persistent-workgroup design: 16 blocks x 384 threads (12 wave32).
// Block b owns batch rows [16b, 16b+16). Each wave owns 2 gate-column tiles
// (w and w+12 of 24). All weight WMMA B-fragments stay in VGPRs for the whole
// T loop (~208 VGPRs @ 3 waves/SIMD -> no spills). Hidden state lives in LDS.
// WMMA chains start from C=0 (inline) and biases are folded into the gate
// nonlinearity, so no accumulator-init vectors stay live across the loop.
// x_t is double-buffered in LDS: loads for t+1 are issued at the top of step
// t and hide a full step of latency (no staging barrier needed).
// ---------------------------------------------------------------------------

typedef __bf16  v16bf __attribute__((ext_vector_type(16)));
typedef float   v8f   __attribute__((ext_vector_type(8)));

#define Bsz 256
#define Tn  4096
#define In  26
#define Hn  128
#define On  26
#define NTHREADS 384   // 12 wave32

union FragU { v16bf v; unsigned u[8]; };

__device__ __forceinline__ v8f wmma_bf16(v16bf a, v16bf b, v8f c) {
    return __builtin_amdgcn_wmma_f32_16x16x32_bf16(
        /*neg_a=*/false, a, /*neg_b=*/false, b,
        /*c_mod=*/(short)0, c, /*reuse_a=*/false, /*reuse_b=*/false);
}

__device__ __forceinline__ float fsig(float x) {
    return 1.0f / (1.0f + __expf(-x));
}
__device__ __forceinline__ float ftanh(float x) {
    return 1.0f - 2.0f / (1.0f + __expf(2.0f * x));
}

// A-matrix fragment (16x32 bf16, M = row = lane%16) from row-major LDS.
// Per ISA 7.12.2: lanes 0-15 -> VGPR0-3: K=0..7, VGPR4-7: K=16..23;
//                 lanes 16-31 -> VGPR0-3: K=8..15, VGPR4-7: K=24..31.
__device__ __forceinline__ v16bf load_A(const __bf16* base, int stride,
                                        int row, int kbase, int half) {
    const unsigned* p = (const unsigned*)(base + row * stride);
    int b0 = (kbase + 8 * half) >> 1;   // pair index, VGPR0..3
    int b1 = b0 + 8;                    // (kbase + 16 + 8*half)/2
    FragU f;
    f.u[0] = p[b0];     f.u[1] = p[b0 + 1];
    f.u[2] = p[b0 + 2]; f.u[3] = p[b0 + 3];
    f.u[4] = p[b1];     f.u[5] = p[b1 + 1];
    f.u[6] = p[b1 + 2]; f.u[7] = p[b1 + 3];
    return f.v;
}

// B-matrix fragment (32x16 bf16, N = col = lane%16) built once from global
// f32 weights W[g][k] (row-major [Gdim][Kdim]); D = A * W^T, so B[k][n]=W[g][k].
// Lanes 0-15 hold K=+0..7,+16..23; lanes 16-31 hold K=+8..15,+24..31 (matches load_A).
__device__ __forceinline__ v16bf load_B(const float* __restrict__ W, int g,
                                        int Kdim, int kbase, int half) {
    FragU f;
    int kb = kbase + 8 * half;
#pragma unroll
    for (int i = 0; i < 8; ++i) {
        int k0 = kb + 2 * i + ((i >= 4) ? 8 : 0);
        float a = (k0     < Kdim) ? W[g * Kdim + k0]     : 0.0f;
        float b = (k0 + 1 < Kdim) ? W[g * Kdim + k0 + 1] : 0.0f;
        __bf16 ba = (__bf16)a, bb = (__bf16)b;
        unsigned short ua, ub;
        __builtin_memcpy(&ua, &ba, 2);
        __builtin_memcpy(&ub, &bb, 2);
        f.u[i] = (unsigned)ua | ((unsigned)ub << 16);
    }
    return f.v;
}

// One GRU layer step. `wave` is an SGPR (readfirstlane), so every role branch
// is a scalar branch: EXEC stays all-ones around each WMMA.
// Tiles 0-7 = r, 8-15 = z, 16-23 = n. Wave w owns tiles w and w+12, so only
// the rep==1 tile can be an n-tile (w>=4) -> one (nx,nh) pair lives across
// the first barrier. WMMA chains start at C=0; biases added post-hoc.
#define LAYER_STEP(XIN, XSTRIDE, XKS, FX, FH, BC, BXN, BHN, HF, HBF)           \
    do {                                                                       \
        v8f nx, nh;                                                            \
        _Pragma("unroll") for (int rep = 0; rep < 2; ++rep) {                  \
            const int tile_ = wave + 12 * rep;                                 \
            v8f ax = {}, ah = {};                                              \
            _Pragma("unroll") for (int ks = 0; ks < (XKS); ++ks)               \
                ax = wmma_bf16(                                                \
                    load_A(&XIN[0][0], (XSTRIDE), ncol, 32 * ks, half),        \
                    (FX)[rep][ks], ax);                                        \
            if (tile_ < 16) { /* r or z: fused x+h accumulation */             \
                _Pragma("unroll") for (int ks = 0; ks < 4; ++ks)               \
                    ax = wmma_bf16(                                            \
                        load_A(&HBF[0][0], 136, ncol, 32 * ks, half),          \
                        (FH)[rep][ks], ax);                                    \
                float* dst_ = (tile_ < 8) ? &r_buf[0][0] : &z_buf[0][0];       \
                int c_ = gcol[rep] & 127;                                      \
                _Pragma("unroll") for (int i = 0; i < 8; ++i)                  \
                    dst_[(i + 8 * half) * 128 + c_] =                          \
                        fsig(ax[i] + (BC)[rep]);                               \
            } else { /* n gate: keep x-part and h-part separate */             \
                _Pragma("unroll") for (int ks = 0; ks < 4; ++ks)               \
                    ah = wmma_bf16(                                            \
                        load_A(&HBF[0][0], 136, ncol, 32 * ks, half),          \
                        (FH)[rep][ks], ah);                                    \
                nx = ax; nh = ah;                                              \
            }                                                                  \
        }                                                                      \
        __syncthreads(); /* r,z visible */                                     \
        if (wave >= 4) { /* this wave's rep==1 tile is an n tile */            \
            int c_ = gcol[1] & 127;                                            \
            _Pragma("unroll") for (int i = 0; i < 8; ++i) {                    \
                float rv_ = r_buf[i + 8 * half][c_];                           \
                n_buf[i + 8 * half][c_] =                                      \
                    ftanh((nx[i] + (BXN)) + rv_ * (nh[i] + (BHN)));            \
            }                                                                  \
        }                                                                      \
        __syncthreads(); /* n visible */                                       \
        for (int idx = tid; idx < 16 * 128; idx += NTHREADS) {                 \
            int m_ = idx >> 7, c_ = idx & 127;                                 \
            float zv_ = z_buf[m_][c_], nv_ = n_buf[m_][c_], hv_ = HF[m_][c_];  \
            float hn_ = (1.0f - zv_) * nv_ + zv_ * hv_;                        \
            HF[m_][c_]  = hn_;                                                 \
            HBF[m_][c_] = (__bf16)hn_;                                         \
        }                                                                      \
        __syncthreads(); /* h updated */                                       \
    } while (0)

__global__ __launch_bounds__(NTHREADS, 1)
void gru2_persistent_kernel(const float* __restrict__ x,
                            const float* __restrict__ h0in,
                            const float* __restrict__ Wih0,
                            const float* __restrict__ Whh0,
                            const float* __restrict__ bih0,
                            const float* __restrict__ bhh0,
                            const float* __restrict__ Wih1,
                            const float* __restrict__ Whh1,
                            const float* __restrict__ bih1,
                            const float* __restrict__ bhh1,
                            const float* __restrict__ fcw,
                            const float* __restrict__ fcb,
                            float* __restrict__ out) {
    // Hidden state + staging in LDS (~52 KB of 320 KB)
    __shared__ __bf16 h0_bf[16][136];     // padded stride (8B-aligned rows)
    __shared__ __bf16 h1_bf[16][136];
    __shared__ __bf16 x_bf[2][16][32];    // double-buffered x_t, K 26 -> 32
    __shared__ float  h0_f[16][128];
    __shared__ float  h1_f[16][128];
    __shared__ float  r_buf[16][128];
    __shared__ float  z_buf[16][128];
    __shared__ float  n_buf[16][128];

    const int tid  = threadIdx.x;
    // Scalarize the wave id: role branches become s_cbranch, EXEC stays
    // all-ones around WMMA.
    const int wave = __builtin_amdgcn_readfirstlane(tid >> 5); // 0..11
    const int lane = tid & 31;
    const int half = lane >> 4;
    const int ncol = lane & 15;         // N (B/C/D) or M (A) index
    const int bb   = blockIdx.x << 4;   // first batch row of this block

    // This lane's gate columns for its two tiles
    int gcol[2];
    gcol[0] = (wave << 4) + ncol;            // tile w
    gcol[1] = ((wave + 12) << 4) + ncol;     // tile w+12

    // ---- Preload all weight B-fragments into VGPRs (bf16), once ----
    v16bf fW0ih[2][1], fW0hh[2][4], fW1ih[2][4], fW1hh[2][4];
    float bc0[2], bc1[2];
#pragma unroll
    for (int rep = 0; rep < 2; ++rep) {
        const int g = gcol[rep];
        fW0ih[rep][0] = load_B(Wih0, g, In, 0, half);   // K=26 padded to 32
#pragma unroll
        for (int ks = 0; ks < 4; ++ks) {
            fW0hh[rep][ks] = load_B(Whh0, g, Hn, 32 * ks, half);
            fW1ih[rep][ks] = load_B(Wih1, g, Hn, 32 * ks, half);
            fW1hh[rep][ks] = load_B(Whh1, g, Hn, 32 * ks, half);
        }
        bc0[rep] = bih0[g] + bhh0[g];   // fused r/z bias (unused for n tiles)
        bc1[rep] = bih1[g] + bhh1[g];
    }
    // n-tile split biases (only rep==1 can be an n tile)
    const float bxn0 = bih0[gcol[1]], bhn0 = bhh0[gcol[1]];
    const float bxn1 = bih1[gcol[1]], bhn1 = bhh1[gcol[1]];

    // ---- Per-thread x-staging slots (slot s covers element s of 16x32) ----
    const int sA = tid, sB = tid + NTHREADS;            // sB < 512 iff tid<128
    const int mA = sA >> 5, cA = sA & 31;
    const int mB = sB >> 5, cB = sB & 31;
    const bool vA = (cA < In);
    const bool vB = (sB < 512) && (cB < In);
    const float* pA = x + (size_t)(bb + mA) * Tn * In + cA;
    const float* pB = x + (size_t)(bb + (mB & 15)) * Tn * In + cB;
    __bf16* lA0 = &x_bf[0][mA][cA];      __bf16* lA1 = &x_bf[1][mA][cA];
    __bf16* lB0 = &x_bf[0][mB & 15][cB]; __bf16* lB1 = &x_bf[1][mB & 15][cB];

    // ---- Prologue: zero x pad, init hidden state, stage x_0 ----
    for (int idx = tid; idx < 2 * 16 * 32; idx += NTHREADS)
        ((__bf16*)x_bf)[idx] = (__bf16)0.0f;
    for (int idx = tid; idx < 16 * 128; idx += NTHREADS) {
        int m = idx >> 7, c = idx & 127;
        float v0 = h0in[(0 * Bsz + bb + m) * Hn + c];
        float v1 = h0in[(1 * Bsz + bb + m) * Hn + c];
        h0_f[m][c] = v0; h0_bf[m][c] = (__bf16)v0;
        h1_f[m][c] = v1; h1_bf[m][c] = (__bf16)v1;
    }
    __syncthreads();
    if (vA) *lA0 = (__bf16)pA[0];
    if (vB) *lB0 = (__bf16)pB[0];
    __syncthreads();

    // ---- Sequential scan over time, both layers pipelined per step ----
    for (int t = 0; t < Tn; ++t) {
        // Prefetch x_{t+1} into the idle buffer: its readers are separated
        // from these writes by this step's barriers -> no extra barrier.
        if (t + 1 < Tn) {
            const size_t off = (size_t)(t + 1) * In;
            if (vA) { float v = pA[off]; *((t & 1) ? lA0 : lA1) = (__bf16)v; }
            if (vB) { float v = pB[off]; *((t & 1) ? lB0 : lB1) = (__bf16)v; }
        }
        // Layer 0: input = x_t (1 K-step), hidden = h0
        LAYER_STEP(x_bf[t & 1], 32, 1, fW0ih, fW0hh, bc0, bxn0, bhn0,
                   h0_f, h0_bf);
        // Layer 1: input = h0 (just updated, 4 K-steps), hidden = h1
        LAYER_STEP(h0_bf, 136, 4, fW1ih, fW1hh, bc1, bxn1, bhn1,
                   h1_f, h1_bf);
    }

    // ---- FC epilogue on last hidden state: [16,128] x [128,26] + bias ----
    for (int idx = tid; idx < 16 * On; idx += NTHREADS) {
        int m = idx / On, o = idx % On;
        float s = fcb[o];
#pragma unroll 8
        for (int k = 0; k < Hn; ++k)
            s += h1_f[m][k] * fcw[o * Hn + k];
        out[(bb + m) * On + o] = s;
    }
}

extern "C" void kernel_launch(void* const* d_in, const int* in_sizes, int n_in,
                              void* d_out, int out_size, void* d_ws, size_t ws_size,
                              hipStream_t stream) {
    (void)in_sizes; (void)n_in; (void)d_ws; (void)ws_size; (void)out_size;
    const float* x    = (const float*)d_in[0];
    const float* h0   = (const float*)d_in[1];
    const float* Wih0 = (const float*)d_in[2];
    const float* Whh0 = (const float*)d_in[3];
    const float* bih0 = (const float*)d_in[4];
    const float* bhh0 = (const float*)d_in[5];
    const float* Wih1 = (const float*)d_in[6];
    const float* Whh1 = (const float*)d_in[7];
    const float* bih1 = (const float*)d_in[8];
    const float* bhh1 = (const float*)d_in[9];
    const float* fcw  = (const float*)d_in[10];
    const float* fcb  = (const float*)d_in[11];
    float* out = (float*)d_out;

    dim3 grid(Bsz / 16);          // 16 persistent workgroups (batch tiles)
    dim3 block(NTHREADS);         // 12 wave32 per workgroup
    gru2_persistent_kernel<<<grid, block, 0, stream>>>(
        x, h0, Wih0, Whh0, bih0, bhh0, Wih1, Whh1, bih1, bhh1, fcw, fcb, out);
}
